// GraphWaveNet_20950850470512
// MI455X (gfx1250) — matrix operand order repaired and checked by reading, not compiled, not run on hardware
//
#include <hip/hip_runtime.h>
#include <hip/hip_bf16.h>

// ---------------------------------------------------------------------------
// GraphWaveNet forward on MI455X (gfx1250, wave32).
// Dominant cost: 832 x [1024x1024]@[1024x32] dense-adjacency GEMMs (~56 GFLOP)
// -> v_wmma_f32_16x16x32_f16 with f32 accumulation; adjacency pre-swizzled to
// the CDNA5 16-bit A-fragment layout once (2MB, L2-resident). Y activations
// staged via LDS in 4x256-row chunks so only 16 B fragments (128 VGPRs) get
// register-hoisted -> higher occupancy to hide the L2 latency of streamed A.
// ---------------------------------------------------------------------------

typedef _Float16 half_t;
typedef _Float16 v16h __attribute__((ext_vector_type(16)));
typedef float    v8f  __attribute__((ext_vector_type(8)));

#define NNODES 1024
#define TMAX   13
#define BATCH  16
#define RES    32
#define SKIPC  256
#define ENDC   512
#define OUTT   12

// ------------------------- input 1x1 conv ---------------------------------
__global__ __launch_bounds__(256) void k_h0(const float* __restrict__ x,
                                            const float* __restrict__ in_w,
                                            const float* __restrict__ in_b,
                                            float* __restrict__ h) {
  int i = blockIdx.x * 256 + threadIdx.x;
  int total = BATCH * TMAX * NNODES * RES;
  if (i >= total) return;
  int c = i & 31;
  int n = (i >> 5) & 1023;
  int bt = i >> 15;                 // b*13 + t
  size_t xb = ((size_t)bt * NNODES + n) * 2;
  h[i] = in_w[c * 2 + 0] * x[xb] + in_w[c * 2 + 1] * x[xb + 1] + in_b[c];
}

// --------------- adaptive adjacency: softmax(relu(e1@e2)) ------------------
__global__ __launch_bounds__(256) void k_adp(const float* __restrict__ e1,
                                             const float* __restrict__ e2,
                                             float* __restrict__ adj,
                                             float* __restrict__ degA) {
  __shared__ float red[256];
  __shared__ float e1s[10];
  int i = blockIdx.x, tid = threadIdx.x;
  if (tid < 10) e1s[tid] = e1[i * 10 + tid];
  __syncthreads();
  float lm = -1e30f;
  for (int j = tid; j < NNODES; j += 256) {
    float m = 0.f;
    for (int e = 0; e < 10; ++e) m += e1s[e] * e2[e * NNODES + j];
    m = fmaxf(m, 0.f);
    lm = fmaxf(lm, m);
  }
  red[tid] = lm; __syncthreads();
  for (int s = 128; s > 0; s >>= 1) {
    if (tid < s) red[tid] = fmaxf(red[tid], red[tid + s]);
    __syncthreads();
  }
  float mx = red[0]; __syncthreads();
  float lsum = 0.f;
  for (int j = tid; j < NNODES; j += 256) {
    float m = 0.f;
    for (int e = 0; e < 10; ++e) m += e1s[e] * e2[e * NNODES + j];
    m = fmaxf(m, 0.f);
    float ex = __expf(m - mx);
    adj[(size_t)i * NNODES + j] = ex;
    lsum += ex;
  }
  red[tid] = lsum; __syncthreads();
  for (int s = 128; s > 0; s >>= 1) {
    if (tid < s) red[tid] += red[tid + s];
    __syncthreads();
  }
  float inv = 1.f / red[0]; __syncthreads();
  float ldeg = 0.f;
  for (int j = tid; j < NNODES; j += 256) {
    float p = adj[(size_t)i * NNODES + j] * inv;
    if (j == i) p = 1.f;            // diag set AFTER softmax (matches ref)
    adj[(size_t)i * NNODES + j] = p;
    ldeg += p;
  }
  red[tid] = ldeg; __syncthreads();
  for (int s = 128; s > 0; s >>= 1) {
    if (tid < s) red[tid] += red[tid + s];
    __syncthreads();
  }
  if (tid == 0) degA[i] = red[0];
}

__global__ __launch_bounds__(256) void k_dinv(const float* __restrict__ degA,
                                              float* __restrict__ dinv) {
  int i = blockIdx.x * 256 + threadIdx.x;
  if (i < NNODES) dinv[i] = rsqrtf(fmaxf(degA[i], 1.0f));
}

// Swizzle normalized adjacency into the CDNA5 16-bit A-fragment layout (f16).
// Layout [mt(64)][kt(32)][lane(32)][16 halves]: lane L -> row M = mt*16+(L&15);
// element i: v=i/2, h=i&1; k = (v<4 ? 2v : 16+2(v-4)) + h + (L>=16 ? 8 : 0).
__global__ void k_swz(const float* __restrict__ adj,
                      const float* __restrict__ dinv,
                      half_t* __restrict__ ah) {
  int blk = blockIdx.x;             // mt*32 + kt
  int kt = blk & 31;
  int lane = threadIdx.x;
  int row = (blk >> 5) * 16 + (lane & 15);
  int hi = lane >> 4;
  float di = dinv[row];
  half_t* dst = ah + ((size_t)blk * 32 + lane) * 16;
  for (int i = 0; i < 16; ++i) {
    int v = i >> 1, hb = i & 1;
    int kl = (v < 4 ? 2 * v : 16 + 2 * (v - 4)) + hb + (hi ? 8 : 0);
    int j = kt * 32 + kl;
    dst[i] = (half_t)(di * adj[(size_t)row * NNODES + j] * dinv[j]);
  }
}

// ------------------ static graph norm + CSR (by destination) ---------------
__global__ __launch_bounds__(256) void k_deg(const int* __restrict__ ei,
                                             const float* __restrict__ ew, int E,
                                             float* __restrict__ degS,
                                             int* __restrict__ cnt) {
  int i = blockIdx.x * 256 + threadIdx.x;
  if (i >= E + NNODES) return;
  int dst; float w;
  if (i < E) { dst = ei[E + i]; w = ew[i]; }
  else       { dst = i - E;     w = 1.0f;  }
  atomicAdd(&degS[dst], w);
  atomicAdd(&cnt[dst], 1);
}

__global__ __launch_bounds__(256) void k_dis(const float* __restrict__ degS,
                                             float* __restrict__ dis) {
  int i = blockIdx.x * 256 + threadIdx.x;
  if (i < NNODES) dis[i] = degS[i] > 0.f ? rsqrtf(degS[i]) : 0.f;
}

__global__ __launch_bounds__(1024) void k_scan(const int* __restrict__ cnt,
                                               int* __restrict__ rp) {
  __shared__ int s[1024];
  int tid = threadIdx.x;
  s[tid] = cnt[tid];
  __syncthreads();
  for (int off = 1; off < 1024; off <<= 1) {
    int v = (tid >= off) ? s[tid - off] : 0;
    __syncthreads();
    s[tid] += v;
    __syncthreads();
  }
  rp[tid + 1] = s[tid];
  if (tid == 0) rp[0] = 0;
}

__global__ __launch_bounds__(256) void k_fill(const int* __restrict__ ei,
                                              const float* __restrict__ ew,
                                              const float* __restrict__ dis,
                                              const int* __restrict__ rp,
                                              int* __restrict__ cur, int E,
                                              int* __restrict__ cols,
                                              float* __restrict__ val) {
  int i = blockIdx.x * 256 + threadIdx.x;
  if (i >= E + NNODES) return;
  int src, dst; float w;
  if (i < E) { src = ei[i]; dst = ei[E + i]; w = ew[i]; }
  else       { src = dst = i - E;            w = 1.0f;  }
  int pos = atomicAdd(&cur[dst], 1);
  int idx = rp[dst] + pos;
  cols[idx] = src;
  val[idx] = dis[src] * w * dis[dst];
}

// ------------- gated TCN + right-side transforms (per layer) ---------------
__global__ __launch_bounds__(256) void k_gate(const float* __restrict__ hin,
                                              const float* __restrict__ tAw,
                                              const float* __restrict__ tAb,
                                              const float* __restrict__ tBw,
                                              const float* __restrict__ tBb,
                                              const float* __restrict__ gw,
                                              const float* __restrict__ gaw,
                                              float* __restrict__ xw,
                                              half_t* __restrict__ yh,
                                              float* __restrict__ glast,
                                              int k, int d, int Tk) {
  __shared__ float Wa0[1024], Wa1[1024], Wb0[1024], Wb1[1024], Wg[1024], Wad[1024];
  __shared__ float hs0[256], hs1[256], gs[256];
  int tid = threadIdx.x;
  for (int i = tid; i < 1024; i += 256) {
    size_t wb = (size_t)(k * 32) * 32 + i;      // [o][ci] or [ci][co] flat
    Wa0[i] = tAw[wb * 2 + 0]; Wa1[i] = tAw[wb * 2 + 1];
    Wb0[i] = tBw[wb * 2 + 0]; Wb1[i] = tBw[wb * 2 + 1];
    Wg[i]  = gw[wb];
    Wad[i] = gaw[wb];
  }
  int id = blockIdx.x;
  int n8 = id & 127;
  int t  = (id >> 7) % Tk;
  int b  = id / (Tk << 7);
  int nl = tid >> 5, c = tid & 31;
  int n = n8 * 8 + nl;
  hs0[tid] = hin[(((size_t)b * TMAX + t) * NNODES + n) * 32 + c];
  hs1[tid] = hin[(((size_t)b * TMAX + t + d) * NNODES + n) * 32 + c];
  __syncthreads();
  float a = tAb[k * 32 + c], bg = tBb[k * 32 + c];
  for (int ci = 0; ci < 32; ++ci) {
    float h0 = hs0[nl * 32 + ci], h1 = hs1[nl * 32 + ci];
    a  += Wa0[c * 32 + ci] * h0 + Wa1[c * 32 + ci] * h1;
    bg += Wb0[c * 32 + ci] * h0 + Wb1[c * 32 + ci] * h1;
  }
  float g = tanhf(a) * (1.f / (1.f + __expf(-bg)));
  gs[nl * 32 + c] = g;
  if (t == Tk - 1) glast[((size_t)b * NNODES + n) * 32 + c] = g;
  __syncthreads();
  float xv = 0.f, yv = 0.f;
  for (int ci = 0; ci < 32; ++ci) {
    float gg = gs[nl * 32 + ci];
    xv += gg * Wg[ci * 32 + c];    // g @ gcn_w
    yv += gg * Wad[ci * 32 + c];   // g @ gcn_adp_w  (-> f16 for WMMA B)
  }
  size_t ob = (((size_t)b * Tk + t) * NNODES + n) * 32 + c;
  xw[ob] = xv;
  yh[ob] = (half_t)yv;
}

// skip conv only needed at last time step of each layer (trailing alignment)
__global__ __launch_bounds__(256) void k_skip(const float* __restrict__ glast,
                                              const float* __restrict__ sw,
                                              const float* __restrict__ sb,
                                              float* __restrict__ skipacc, int k) {
  __shared__ float g[32];
  int bn = blockIdx.x;
  int tid = threadIdx.x;
  if (tid < 32) g[tid] = glast[(size_t)bn * 32 + tid];
  __syncthreads();
  float acc = sb[k * 256 + tid];
  const float* w = sw + ((size_t)k * 256 + tid) * 32;
  for (int c = 0; c < 32; ++c) acc += w[c] * g[c];
  skipacc[(size_t)bn * 256 + tid] += acc;
}

// sparse GCN gather + biases + residual into hh (= hout)
__global__ __launch_bounds__(256) void k_gcn1(const float* __restrict__ hin,
                                              const float* __restrict__ xw,
                                              const int* __restrict__ rp,
                                              const int* __restrict__ cols,
                                              const float* __restrict__ val,
                                              const float* __restrict__ gb,
                                              const float* __restrict__ gab,
                                              float* __restrict__ hout,
                                              int k, int d, int Tk) {
  int id = blockIdx.x;
  int n8 = id & 127;
  int t  = (id >> 7) % Tk;
  int b  = id / (Tk << 7);
  int tid = threadIdx.x;
  int nl = tid >> 5, c = tid & 31;
  int n = n8 * 8 + nl;
  float acc = gb[k * 32 + c] + gab[k * 32 + c]
            + hin[(((size_t)b * TMAX + t + d) * NNODES + n) * 32 + c];
  int e0 = rp[n], e1 = rp[n + 1];
  size_t xb = (((size_t)b * Tk + t) * NNODES) * 32;
  for (int e = e0; e < e1; ++e)
    acc += val[e] * xw[xb + (size_t)cols[e] * 32 + c];
  hout[(((size_t)b * TMAX + t) * NNODES + n) * 32 + c] = acc;
}

// ----------------- dense adaptive GCN via WMMA (the hot loop) --------------
// hh[b,t] += adjn(f16,swizzled) @ Y(f16); fused BatchNorm sum/sumsq.
// 4 K-stages of 256 rows: only 16 B fragments get hoisted to VGPRs (128 regs),
// keeping total pressure low enough for ~2x the occupancy of a 2-stage split.
// A fragments use a 2-deep pipeline with two live values (distinct registers).
__global__ __launch_bounds__(256) void k_wmma(const half_t* __restrict__ ah,
                                              const half_t* __restrict__ yh,
                                              float* __restrict__ hh,
                                              float* __restrict__ bns, int Tk) {
  __shared__ __align__(16) half_t ly[256 * 40];   // 20KB, padded stride
  int bt = blockIdx.x;              // b*Tk + t
  int b = bt / Tk, t = bt % Tk;
  int tid = threadIdx.x;
  int wave = tid >> 5, lane = tid & 31;
  const half_t* ybase = yh + (size_t)bt * NNODES * 32;
  int c = lane & 15;
  int hi = lane >> 4;
  float s0 = 0.f, ss0 = 0.f, s1 = 0.f, ss1 = 0.f;
  for (int jblk = 0; jblk < 4; ++jblk) {
    for (int it = tid; it < 256 * 2; it += 256) {    // stage 256 K-rows
      int row = it >> 1, hseg = it & 1;
      const uint4* src = (const uint4*)(ybase + (size_t)(jblk * 256 + row) * 32 + hseg * 16);
      *(uint4*)&ly[row * 40 + hseg * 16] = *src;
    }
    __syncthreads();
    bool last = (jblk == 3);
    for (int mt = wave; mt < 64; mt += 8) {          // wave-uniform
      v8f acc0 = {}; v8f acc1 = {};
      // per-(mt, jblk) A base: consecutive kt tiles are 512 halves apart
      const half_t* abase = ah + ((size_t)((mt * 32 + jblk * 8) * 32 + lane)) * 16;
      v16h a0 = *(const v16h*)(abase);
      v16h a1 = *(const v16h*)(abase + 512);
      #pragma unroll
      for (int kt = 0; kt < 8; kt += 2) {
        __builtin_prefetch(abase + (size_t)(kt + 4) * 512, 0, 1);
        const half_t* l0 = &ly[(kt * 32) * 40];   // lane = N col, elems = K (table order)
        v16h b00, b01;
        #pragma unroll
        for (int s = 0; s < 16; ++s) {
          int kk = (s < 8 ? 8 * hi + s : 16 + 8 * hi + (s - 8));
          b00[s] = l0[kk * 40 + c];
          b01[s] = l0[kk * 40 + 16 + c];
        }
        acc0 = __builtin_amdgcn_wmma_f32_16x16x32_f16(false, a0, false, b00,
                                                      (short)0, acc0, false, false);
        acc1 = __builtin_amdgcn_wmma_f32_16x16x32_f16(false, a0, false, b01,
                                                      (short)0, acc1, false, false);
        a0 = *(const v16h*)(abase + (size_t)((kt + 2) & 7) * 512);
        const half_t* l1 = &ly[((kt + 1) * 32) * 40];
        v16h b10, b11;
        #pragma unroll
        for (int s = 0; s < 16; ++s) {
          int kk = (s < 8 ? 8 * hi + s : 16 + 8 * hi + (s - 8));
          b10[s] = l1[kk * 40 + c];
          b11[s] = l1[kk * 40 + 16 + c];
        }
        acc0 = __builtin_amdgcn_wmma_f32_16x16x32_f16(false, a1, false, b10,
                                                      (short)0, acc0, false, false);
        acc1 = __builtin_amdgcn_wmma_f32_16x16x32_f16(false, a1, false, b11,
                                                      (short)0, acc1, false, false);
        a1 = *(const v16h*)(abase + (size_t)((kt + 3) & 7) * 512);
      }
      #pragma unroll
      for (int r = 0; r < 8; ++r) {   // C/D layout: VGPR r -> M = r + hi*8
        int row = mt * 16 + hi * 8 + r;
        size_t base = (((size_t)b * TMAX + t) * NNODES + row) * 32;
        float v0 = hh[base + c] + acc0[r];
        float v1 = hh[base + 16 + c] + acc1[r];
        hh[base + c] = v0;
        hh[base + 16 + c] = v1;
        if (last) { s0 += v0; ss0 += v0 * v0; s1 += v1; ss1 += v1 * v1; }
      }
    }
    __syncthreads();
  }
  atomicAdd(&bns[c], s0);        atomicAdd(&bns[32 + c], ss0);
  atomicAdd(&bns[16 + c], s1);   atomicAdd(&bns[32 + 16 + c], ss1);
}

// training-mode BatchNorm (biased var), in place
__global__ __launch_bounds__(256) void k_bn(float* __restrict__ h,
                                            const float* __restrict__ bns,
                                            const float* __restrict__ bng,
                                            const float* __restrict__ bnb,
                                            int k, int Tk) {
  int i = blockIdx.x * 256 + threadIdx.x;
  int total = BATCH * Tk * NNODES * RES;
  if (i >= total) return;
  int c = i & 31;
  int n = (i >> 5) & 1023;
  int bt = i >> 15;
  int t = bt % Tk, b = bt / Tk;
  float cntf = (float)(BATCH * Tk * NNODES);
  float mu = bns[c] / cntf;
  float var = bns[32 + c] / cntf - mu * mu;
  float a = bng[k * 32 + c] * rsqrtf(var + 1e-5f);
  size_t idx = (((size_t)b * TMAX + t) * NNODES + n) * 32 + c;
  h[idx] = (h[idx] - mu) * a + bnb[k * 32 + c];
}

// ------------------------- output head ------------------------------------
__global__ __launch_bounds__(256) void k_end(const float* __restrict__ skipacc,
                                             const float* __restrict__ w1,
                                             const float* __restrict__ b1,
                                             const float* __restrict__ w2,
                                             const float* __restrict__ b2,
                                             float* __restrict__ out) {
  __shared__ float s[256];
  __shared__ float y1[512];
  int bn = blockIdx.x;
  int b = bn >> 10, n = bn & 1023;
  int tid = threadIdx.x;
  s[tid] = fmaxf(skipacc[(size_t)bn * 256 + tid], 0.f);
  __syncthreads();
  for (int rep = 0; rep < 2; ++rep) {
    int o = tid + rep * 256;
    float acc = b1[o];
    const float* w = w1 + (size_t)o * 256;
    for (int c = 0; c < 256; ++c) acc += w[c] * s[c];
    y1[o] = fmaxf(acc, 0.f);
  }
  __syncthreads();
  if (tid < OUTT) {
    float acc = b2[tid];
    const float* w = w2 + (size_t)tid * 512;
    for (int c = 0; c < 512; ++c) acc += w[c] * y1[c];
    out[((size_t)b * OUTT + tid) * NNODES + n] = acc;
  }
}

// ---------------------------------------------------------------------------
extern "C" void kernel_launch(void* const* d_in, const int* in_sizes, int n_in,
                              void* d_out, int out_size, void* d_ws, size_t ws_size,
                              hipStream_t stream) {
  const float* x   = (const float*)d_in[0];
  const int*   ei  = (const int*)d_in[1];
  const float* ew  = (const float*)d_in[2];
  const float* e1  = (const float*)d_in[3];
  const float* e2  = (const float*)d_in[4];
  const float* inw = (const float*)d_in[5];
  const float* inb = (const float*)d_in[6];
  const float* tAw = (const float*)d_in[7];
  const float* tAb = (const float*)d_in[8];
  const float* tBw = (const float*)d_in[9];
  const float* tBb = (const float*)d_in[10];
  const float* gw  = (const float*)d_in[11];
  const float* gb  = (const float*)d_in[12];
  const float* gaw = (const float*)d_in[13];
  const float* gab = (const float*)d_in[14];
  const float* sw  = (const float*)d_in[15];
  const float* sb  = (const float*)d_in[16];
  const float* bng = (const float*)d_in[17];
  const float* bnb = (const float*)d_in[18];
  const float* w1  = (const float*)d_in[19];
  const float* b1  = (const float*)d_in[20];
  const float* w2  = (const float*)d_in[21];
  const float* b2  = (const float*)d_in[22];
  (void)n_in; (void)out_size; (void)ws_size;

  int E  = in_sizes[2];            // edge_weight element count
  int EN = E + NNODES;

  char* ws = (char*)d_ws;
  size_t off = 0;
  auto alloc = [&](size_t bytes) -> void* {
    void* p = ws + off;
    off += (bytes + 255) & ~(size_t)255;
    return p;
  };
  float*  adj   = (float*)alloc(4ull * NNODES * NNODES);
  float*  degA  = (float*)alloc(4 * NNODES);
  float*  dinv  = (float*)alloc(4 * NNODES);
  half_t* ah    = (half_t*)alloc(2ull * NNODES * NNODES);
  float*  degS  = (float*)alloc(4 * NNODES);
  float*  dis   = (float*)alloc(4 * NNODES);
  int*    cnt   = (int*)alloc(4 * NNODES);
  int*    rp    = (int*)alloc(4 * (NNODES + 1));
  int*    cur   = (int*)alloc(4 * NNODES);
  int*    cols  = (int*)alloc(4ull * EN);
  float*  val   = (float*)alloc(4ull * EN);
  float*  hA    = (float*)alloc(4ull * BATCH * TMAX * NNODES * RES);
  float*  hB    = (float*)alloc(4ull * BATCH * TMAX * NNODES * RES);
  float*  xwbuf = (float*)alloc(4ull * BATCH * 12 * NNODES * RES);
  half_t* yh    = (half_t*)alloc(2ull * BATCH * 12 * NNODES * RES);
  float*  glast = (float*)alloc(4ull * BATCH * NNODES * RES);
  float*  skipa = (float*)alloc(4ull * BATCH * NNODES * SKIPC);
  float*  bns   = (float*)alloc(4 * 64);

  hipMemsetAsync(skipa, 0, 4ull * BATCH * NNODES * SKIPC, stream);
  hipMemsetAsync(degS, 0, 4 * NNODES, stream);
  hipMemsetAsync(cnt, 0, 4 * NNODES, stream);
  hipMemsetAsync(cur, 0, 4 * NNODES, stream);

  {
    int total = BATCH * TMAX * NNODES * RES;
    k_h0<<<(total + 255) / 256, 256, 0, stream>>>(x, inw, inb, hA);
  }
  k_adp<<<NNODES, 256, 0, stream>>>(e1, e2, adj, degA);
  k_dinv<<<NNODES / 256, 256, 0, stream>>>(degA, dinv);
  k_swz<<<64 * 32, 32, 0, stream>>>(adj, dinv, ah);
  k_deg<<<(EN + 255) / 256, 256, 0, stream>>>(ei, ew, E, degS, cnt);
  k_dis<<<NNODES / 256, 256, 0, stream>>>(degS, dis);
  k_scan<<<1, 1024, 0, stream>>>(cnt, rp);
  k_fill<<<(EN + 255) / 256, 256, 0, stream>>>(ei, ew, dis, rp, cur, E, cols, val);

  const int dil[8] = {1, 2, 1, 2, 1, 2, 1, 2};
  float* hin = hA;
  float* hout = hB;
  int Tin = TMAX;
  for (int k = 0; k < 8; ++k) {
    int d = dil[k];
    int Tk = Tin - d;
    hipMemsetAsync(bns, 0, 4 * 64, stream);
    int blocks = BATCH * Tk * (NNODES / 8);
    k_gate<<<blocks, 256, 0, stream>>>(hin, tAw, tAb, tBw, tBb, gw, gaw,
                                       xwbuf, yh, glast, k, d, Tk);
    k_skip<<<BATCH * NNODES, 256, 0, stream>>>(glast, sw, sb, skipa, k);
    k_gcn1<<<blocks, 256, 0, stream>>>(hin, xwbuf, rp, cols, val, gb, gab,
                                       hout, k, d, Tk);
    k_wmma<<<BATCH * Tk, 256, 0, stream>>>(ah, yh, hout, bns, Tk);
    int total = BATCH * Tk * NNODES * RES;
    k_bn<<<(total + 255) / 256, 256, 0, stream>>>(hout, bns, bng, bnb, k, Tk);
    float* tmp = hin; hin = hout; hout = tmp;
    Tin = Tk;
  }
  k_end<<<BATCH * NNODES, 256, 0, stream>>>(skipa, w1, b1, w2, b2, (float*)d_out);
}
